// HungarianPredictor_18442589569644
// MI455X (gfx1250) — matrix operand order repaired and checked by reading, not compiled, not run on hardware
//
#include <hip/hip_runtime.h>

// CDNA5 / gfx1250, wave32. One wave per batch item.
typedef __attribute__((ext_vector_type(16))) _Float16 v16h;
typedef __attribute__((ext_vector_type(8)))  _Float16 v8h;
typedef __attribute__((ext_vector_type(8)))  float    v8f;

#define NSLOT 24
#define DIM   256
#define LSTR  264        // f16 row stride in LDS (256 + 8 pad -> no bank conflicts)
#define BIGF  1e9f       // matches reference INF sentinel

__global__ __launch_bounds__(32)
void hungarian_predictor_kernel(const float* __restrict__ slots,
                                const float* __restrict__ prev,
                                float* __restrict__ out)
{
    __shared__ alignas(16) _Float16 pnh[32 * LSTR];   // normalized prev (f16)
    __shared__ alignas(16) _Float16 cnh[32 * LSTR];   // normalized curr (f16)
    __shared__ float cost[32 * 32];
    __shared__ float u_lds[32];
    __shared__ int   colind[32];

    const int b    = blockIdx.x;
    const int lane = threadIdx.x;                      // 0..31 (wave32)

    const float* __restrict__ srow = slots + (size_t)b * NSLOT * DIM;
    const float* __restrict__ prow = prev  + (size_t)b * NSLOT * DIM;

    __builtin_prefetch(srow, 0, 3);   // global_prefetch_b8
    __builtin_prefetch(prow, 0, 3);

    // ---------------- Phase 1: L2-normalize rows -> f16 LDS ----------------
    for (int m = 0; m < NSLOT; ++m) {
        {   // prev -> pnh
            const float4 x0 = *(const float4*)(prow + m * DIM + lane * 8);
            const float4 x1 = *(const float4*)(prow + m * DIM + lane * 8 + 4);
            float ss = x0.x*x0.x + x0.y*x0.y + x0.z*x0.z + x0.w*x0.w
                     + x1.x*x1.x + x1.y*x1.y + x1.z*x1.z + x1.w*x1.w;
            for (int o = 16; o > 0; o >>= 1) ss += __shfl_xor(ss, o, 32);
            const float inv = 1.0f / fmaxf(sqrtf(ss), 1e-12f);
            _Float16* d = &pnh[m * LSTR + lane * 8];
            d[0]=(_Float16)(x0.x*inv); d[1]=(_Float16)(x0.y*inv);
            d[2]=(_Float16)(x0.z*inv); d[3]=(_Float16)(x0.w*inv);
            d[4]=(_Float16)(x1.x*inv); d[5]=(_Float16)(x1.y*inv);
            d[6]=(_Float16)(x1.z*inv); d[7]=(_Float16)(x1.w*inv);
        }
        {   // slots -> cnh
            const float4 x0 = *(const float4*)(srow + m * DIM + lane * 8);
            const float4 x1 = *(const float4*)(srow + m * DIM + lane * 8 + 4);
            float ss = x0.x*x0.x + x0.y*x0.y + x0.z*x0.z + x0.w*x0.w
                     + x1.x*x1.x + x1.y*x1.y + x1.z*x1.z + x1.w*x1.w;
            for (int o = 16; o > 0; o >>= 1) ss += __shfl_xor(ss, o, 32);
            const float inv = 1.0f / fmaxf(sqrtf(ss), 1e-12f);
            _Float16* d = &cnh[m * LSTR + lane * 8];
            d[0]=(_Float16)(x0.x*inv); d[1]=(_Float16)(x0.y*inv);
            d[2]=(_Float16)(x0.z*inv); d[3]=(_Float16)(x0.w*inv);
            d[4]=(_Float16)(x1.x*inv); d[5]=(_Float16)(x1.y*inv);
            d[6]=(_Float16)(x1.z*inv); d[7]=(_Float16)(x1.w*inv);
        }
    }
    for (int m = NSLOT; m < 32; ++m) {   // zero padding rows 24..31
        v8h z = {};
        *(v8h*)&pnh[m * LSTR + lane * 8] = z;
        *(v8h*)&cnh[m * LSTR + lane * 8] = z;
    }
    __syncthreads();

    // ---- Phase 2: cost = 1 - pn @ cn^T via v_wmma_f32_16x16x32_f16 ----
    // A (16x32 f16): lanes 0-15 row M, K chunks {kb+0..7, kb+16..23};
    //                lanes 16-31 row M, K chunks {kb+8..15, kb+24..31}
    // B (32x16 f16): lanes 0-15 col N, K chunks {kb+0..7, kb+16..23}; lanes 16-31 {kb+8..15, kb+24..31}
    // C/D (16x16 f32): lane = col N, VGPR r = row r (lanes 0-15) / r+8 (16-31)
    const int half16 = lane & 15;
    const int hi     = (lane >> 4) & 1;
    const int asel   = hi ? 8  : 0;
    const int boff   = hi ? 16 : 0;

    for (int ti = 0; ti < 2; ++ti) {
        for (int tj = 0; tj < 2; ++tj) {
            v8f acc = {};
            const int M = ti * 16 + half16;
            const int N = tj * 16 + half16;
            #pragma unroll
            for (int kb = 0; kb < DIM; kb += 32) {
                union { v16h v; v8h h[2]; } A, Bf;
                A.h[0]  = *(const v8h*)&pnh[M * LSTR + kb + asel];
                A.h[1]  = *(const v8h*)&pnh[M * LSTR + kb + 16 + asel];
                Bf.h[0] = *(const v8h*)&cnh[N * LSTR + kb + asel];
                Bf.h[1] = *(const v8h*)&cnh[N * LSTR + kb + 16 + asel];
                acc = __builtin_amdgcn_wmma_f32_16x16x32_f16(
                        false, A.v, false, Bf.v, (short)0, acc, false, false);
            }
            #pragma unroll
            for (int r = 0; r < 8; ++r) {
                const int row = ti * 16 + r + hi * 8;
                cost[row * 32 + tj * 16 + half16] = 1.0f - acc[r];
            }
        }
    }
    u_lds[lane] = 0.0f;
    __syncthreads();

    // ---------------- Phase 3: Hungarian (e-maxx JV), lane j = column j ----
    float v_j = 0.0f;     // column potential (col 24 = virtual start column)
    int   p_j = -1;       // row assigned to column j (-1 = free)

    for (int i = 0; i < NSLOT; ++i) {
        if (lane == NSLOT) p_j = i;    // p[n] = i
        float minv_j = BIGF;
        int   way_j  = 0;
        int   used_j = 0;
        int   j0     = NSLOT;

        while (true) {
            const int i0 = __shfl(p_j, j0, 32);
            if (i0 < 0) break;                      // reached a free column
            if (lane == j0) used_j = 1;
            const float ui0 = u_lds[i0];
            float cur = BIGF;
            if (lane < NSLOT && !used_j)
                cur = cost[i0 * 32 + lane] - ui0 - v_j;
            if (cur < minv_j) { minv_j = cur; way_j = j0; }

            // argmin over unused real columns (tie-break: lowest index)
            float mval = (lane < NSLOT && !used_j) ? minv_j : BIGF;
            int   midx = lane;
            for (int o = 16; o > 0; o >>= 1) {
                const float ov = __shfl_xor(mval, o, 32);
                const int   oi = __shfl_xor(midx, o, 32);
                if (ov < mval || (ov == mval && oi < midx)) { mval = ov; midx = oi; }
            }
            const float delta = mval;

            if (used_j) {                 // distinct p_j per used column: no race
                u_lds[p_j] += delta;
                v_j -= delta;
            } else {
                minv_j -= delta;
            }
            j0 = midx;
        }
        // augment along alternating path back to virtual column
        int j = j0;
        while (j != NSLOT) {
            const int jp = __shfl(way_j, j, 32);
            const int pv = __shfl(p_j, jp, 32);
            if (lane == j) p_j = pv;
            j = jp;
        }
    }
    if (lane < NSLOT) colind[p_j] = lane;   // col_ind[row p[j]] = j
    __syncthreads();

    // ---------------- Phase 4: permuted row gather/copy --------------------
    for (int i = 0; i < NSLOT; ++i) {
        const int c = colind[i];
        const float4 x0 = *(const float4*)(srow + c * DIM + lane * 8);
        const float4 x1 = *(const float4*)(srow + c * DIM + lane * 8 + 4);
        float* __restrict__ drow = out + ((size_t)b * NSLOT + i) * DIM;
        *(float4*)(drow + lane * 8)     = x0;
        *(float4*)(drow + lane * 8 + 4) = x1;
    }
}

extern "C" void kernel_launch(void* const* d_in, const int* in_sizes, int n_in,
                              void* d_out, int out_size, void* d_ws, size_t ws_size,
                              hipStream_t stream) {
    (void)n_in; (void)out_size; (void)d_ws; (void)ws_size;
    const float* slots = (const float*)d_in[0];
    const float* prev  = (const float*)d_in[1];
    float* out = (float*)d_out;
    const int B = in_sizes[0] / (NSLOT * DIM);   // 2048
    hungarian_predictor_kernel<<<dim3(B), dim3(32), 0, stream>>>(slots, prev, out);
}
